// ReprogrammingLayer_9646496547158
// MI455X (gfx1250) — matrix operand order repaired and checked by reading, not compiled, not run on hardware
//
#include <hip/hip_runtime.h>

// ---------------------------------------------------------------------------
// Types
// ---------------------------------------------------------------------------
typedef __bf16 bf16;
typedef __attribute__((ext_vector_type(16))) __bf16 v16bf;
typedef __attribute__((ext_vector_type(8)))  float  v8f;
typedef __attribute__((ext_vector_type(4)))  float  v4f;
typedef __attribute__((ext_vector_type(4)))  unsigned int u32x4;
typedef __attribute__((ext_vector_type(8)))  int    i32x8;
typedef __attribute__((ext_vector_type(4)))  int    i32x4;

#if defined(__has_builtin)
#if __has_builtin(__builtin_amdgcn_tensor_load_to_lds) && \
    __has_builtin(__builtin_amdgcn_s_wait_tensorcnt)
#define HAS_TDM 1
#endif
#endif
#ifndef HAS_TDM
#define HAS_TDM 0
#endif

__device__ __forceinline__ unsigned int pack2(float a, float b) {
  union { bf16 h[2]; unsigned int u; } x;
  x.h[0] = (bf16)a; x.h[1] = (bf16)b;
  return x.u;
}

// Build a 16x32 (or 32x16) bf16 fragment from two 16-byte segments.
__device__ __forceinline__ v16bf load_frag(const bf16* p0, const bf16* p1) {
  union { v16bf v; u32x4 q[2]; } u;
  u.q[0] = *(const u32x4*)p0;
  u.q[1] = *(const u32x4*)p1;
  return u.v;
}

__device__ __forceinline__ v8f wmma_bf16(v16bf a, v16bf b, v8f c) {
  // D(f32 16x16) = A(16x32 bf16) * B(32x16 bf16) + C
  return __builtin_amdgcn_wmma_f32_16x16x32_bf16(false, a, false, b,
                                                 (short)0, c, false, false);
}

// Same-wave LDS RAW fence (wave-private scratch; no workgroup barrier needed).
__device__ __forceinline__ void lds_fence() {
  asm volatile("s_wait_dscnt 0" ::: "memory");
  __builtin_amdgcn_wave_barrier();
}

#if HAS_TDM
// Low 32 bits of a generic LDS pointer == LDS byte offset (LDS aperture only
// occupies the high dword of the generic address).
__device__ __forceinline__ unsigned lds_off(const void* p) {
  return (unsigned)(unsigned long long)(uintptr_t)p;
}

// Issue one 2D TDM tile load: global (row-major, 2-byte elems) -> LDS.
//   tensor_d0/d1 : tensor extent (elements) in dim0 (fastest) / dim1
//   tile_d0/d1   : tile extent (elements)
//   stride_d0    : elements between consecutive dim1 lines
//   pad_int/amt  : D# pad_interval / pad_amount field codes (LDS row padding)
__device__ __forceinline__ void tdm_load_2d(unsigned lds_addr, const void* gptr,
                                            unsigned tensor_d0, unsigned tensor_d1,
                                            unsigned tile_d0, unsigned tile_d1,
                                            unsigned stride_d0,
                                            unsigned pad_int, unsigned pad_amt)
{
  const unsigned long long ga = (unsigned long long)(uintptr_t)gptr;
  u32x4 g0;
  g0[0] = 1u;                                   // count=1 (valid descriptor)
  g0[1] = lds_addr;                             // lds_addr [63:32]
  g0[2] = (unsigned)(ga & 0xffffffffu);         // global_addr[31:0]
  g0[3] = (unsigned)((ga >> 32) & 0x1ffffffu)   // global_addr[56:32]
        | (2u << 30);                           // type = 2 ("image")
  i32x8 g1;
  g1[0] = (int)((1u << 16)                      // data_size = 2 bytes
        | (1u << 20)                            // pad_enable
        | (pad_int << 22)                       // pad_interval
        | (pad_amt << 25));                     // pad_amount
  g1[1] = (int)((tensor_d0 & 0xffffu) << 16);   // tensor_dim0[15:0] @ bits 63:48
  g1[2] = (int)(((tensor_d0 >> 16) & 0xffffu)   // tensor_dim0[31:16]
        | ((tensor_d1 & 0xffffu) << 16));       // tensor_dim1[15:0]
  g1[3] = (int)(((tensor_d1 >> 16) & 0xffffu)   // tensor_dim1[31:16]
        | ((tile_d0 & 0xffffu) << 16));         // tile_dim0 @ bits 127:112
  g1[4] = (int)(tile_d1 & 0xffffu);             // tile_dim1; tile_dim2 = 0
  g1[5] = (int)stride_d0;                       // tensor_dim0_stride[31:0]
  g1[6] = 0;                                    // stride0[47:32] | stride1[15:0]
  g1[7] = 0;
  const i32x4 gz4 = {};                         // groups 2/3: <=2D tensor
  const i32x8 gz8 = {};                         // 6-arg form: extra group, zeroed
  __builtin_amdgcn_tensor_load_to_lds(g0, g1, gz4, gz4, gz8, 0);
}
#endif

// ---------------------------------------------------------------------------
// Generic bf16-WMMA GEMM:  C[M,N] = A[M,K](f32) * B[K,N](f32) + bias[N]
// Rows in [M, Mpad) are written with the bias value (zero A contribution) so
// downstream kernels can consume padded buffers without guards.
// OUT_MODE: 0 = f32 row-major, 1 = bf16 row-major, 2 = bf16 transposed (ldT)
// Block: 256 threads (8 waves), tile 128x128, BK=32.
// ---------------------------------------------------------------------------
template<int OUT_MODE>
__global__ __launch_bounds__(256)
void gemm_bf16(const float* __restrict__ A, const float* __restrict__ B,
               const float* __restrict__ bias, void* __restrict__ Cout,
               int M, int Mpad, int N, int K, int ldT)
{
  __shared__ bf16 As[128 * 40];   // [row][k], stride 40 bf16 (bank-conflict free)
  __shared__ bf16 Bs[128 * 40];   // [n][k] (transposed), stride 40

  const int tid  = threadIdx.x;
  const int lane = tid & 31;
  const int lm   = lane & 15;
  const int hf   = lane >> 4;
  const int wave = tid >> 5;
  const int wm   = wave & 3;       // 4 row-groups of 32
  const int wn   = wave >> 2;      // 2 col-groups of 64
  const int m0   = blockIdx.y * 128;
  const int n0   = blockIdx.x * 128;

  v8f acc[2][4];
  #pragma unroll
  for (int i = 0; i < 2; i++)
    #pragma unroll
    for (int j = 0; j < 4; j++)
      acc[i][j] = (v8f){};

  const int arow = tid >> 1;          // 0..127
  const int ak   = (tid & 1) * 16;    // 0 or 16
  const int bn   = tid & 127;         // 0..127
  const int bk   = (tid >> 7) * 16;   // 0 or 16

  for (int kk = 0; kk < K; kk += 32) {
    // ---- stage A tile (f32 -> bf16), rows >= M zero-filled ----
    {
      u32x4 w0 = {}, w1 = {};
      const int gm = m0 + arow;
      if (gm < M) {
        const v4f* ap = (const v4f*)(A + (size_t)gm * K + kk + ak);
        v4f f0 = ap[0], f1 = ap[1], f2 = ap[2], f3 = ap[3];
        w0 = (u32x4){pack2(f0[0], f0[1]), pack2(f0[2], f0[3]),
                     pack2(f1[0], f1[1]), pack2(f1[2], f1[3])};
        w1 = (u32x4){pack2(f2[0], f2[1]), pack2(f2[2], f2[3]),
                     pack2(f3[0], f3[1]), pack2(f3[2], f3[3])};
      }
      *(u32x4*)(&As[arow * 40 + ak])     = w0;
      *(u32x4*)(&As[arow * 40 + ak + 8]) = w1;
    }
    // ---- stage B tile transposed: Bs[n][k] <- B[kk+k][n0+n] ----
    {
      const float* bp = B + (size_t)(kk + bk) * N + n0 + bn;
      float f[16];
      #pragma unroll
      for (int j = 0; j < 16; j++) f[j] = bp[(size_t)j * N];
      u32x4 w0 = (u32x4){pack2(f[0], f[1]),  pack2(f[2], f[3]),
                         pack2(f[4], f[5]),  pack2(f[6], f[7])};
      u32x4 w1 = (u32x4){pack2(f[8], f[9]),  pack2(f[10], f[11]),
                         pack2(f[12], f[13]), pack2(f[14], f[15])};
      *(u32x4*)(&Bs[bn * 40 + bk])     = w0;
      *(u32x4*)(&Bs[bn * 40 + bk + 8]) = w1;
    }
    __syncthreads();

    // ---- fragments + 8 WMMAs ----
    v16bf af[2], bfv[4];
    #pragma unroll
    for (int i = 0; i < 2; i++) {
      // A layout: elems0..15 = k(16h..16h+15), M = lane%16
      const bf16* p = &As[(wm * 32 + i * 16 + lm) * 40];
      af[i] = load_frag(p + 16 * hf, p + 16 * hf + 8);
    }
    #pragma unroll
    for (int j = 0; j < 4; j++) {
      // B layout: elems0..15 = k(16h..16h+15), N = lane%16
      const bf16* p = &Bs[(wn * 64 + j * 16 + lm) * 40 + 16 * hf];
      bfv[j] = load_frag(p, p + 8);
    }
    #pragma unroll
    for (int i = 0; i < 2; i++)
      #pragma unroll
      for (int j = 0; j < 4; j++)
        acc[i][j] = wmma_bf16(af[i], bfv[j], acc[i][j]);
    __syncthreads();
  }

  // ---- epilogue: bias + store (rows < Mpad) ----
  #pragma unroll
  for (int j = 0; j < 4; j++) {
    const int col = n0 + wn * 64 + j * 16 + lm;
    const float bb = bias[col];
    #pragma unroll
    for (int i = 0; i < 2; i++) {
      #pragma unroll
      for (int r = 0; r < 8; r++) {
        const int row = m0 + wm * 32 + i * 16 + r + 8 * hf;  // C layout
        if (row < Mpad) {
          const float v = acc[i][j][r] + bb;
          if (OUT_MODE == 0)
            ((float*)Cout)[(size_t)row * N + col] = v;
          else if (OUT_MODE == 1)
            ((bf16*)Cout)[(size_t)row * N + col] = (bf16)v;
          else
            ((bf16*)Cout)[(size_t)col * ldT + row] = (bf16)v;
        }
      }
    }
  }
}

// ---------------------------------------------------------------------------
// Fused flash-attention over one (batch, head, 64-row slab).
// Q: bf16 [B*L,1024], Kb: bf16 [1024(S padded),1024],
// Vt: bf16 [1024,1024] (per-head [E][S_pad] transposed), Out: f32 [B*L,1024].
// Block = 128 threads (4 waves x 16 q-rows).  K/V tiles are streamed into a
// double-buffered LDS pipeline by the Tensor Data Mover (wave 0 issues,
// s_wait_tensorcnt + barrier synchronize); padded keys (>=S) are masked in
// the online softmax, so no staging guards are needed.
// ---------------------------------------------------------------------------
__global__ __launch_bounds__(128)
void attn_fused(const bf16* __restrict__ Q, const bf16* __restrict__ Kb,
                const bf16* __restrict__ Vt, float* __restrict__ Out, int S)
{
  __shared__ bf16  Ks[2][32 * 136];  // [key][e], stride 136 bf16 (272B rows)
  __shared__ bf16  Vs[2][128 * 40];  // [e][key], stride 40 bf16 (80B rows)
  __shared__ float Pw[4][16 * 32];   // per-wave raw scores  [row][key]
  __shared__ bf16  Pb[4][16 * 32];   // per-wave exp(P) bf16 [row][key]
  __shared__ float red[4][16];       // per-wave row broadcast (alpha / l)

  const int tid  = threadIdx.x;
  const int lane = tid & 31;
  const int lm   = lane & 15;
  const int hf   = lane >> 4;
  const int wave = tid >> 5;
  const int bh   = blockIdx.x;          // 0..127 == b*8 + h
  const int hh   = bh & 7;
  const int bb   = bh >> 3;
  const int row0 = bb * 512 + blockIdx.y * 64;
  const int ho   = hh * 128;
  const float scale = 0.08838834764831845f;   // 1/sqrt(128)
  const float NEG = -__builtin_inff();
  const int NCHUNK = 32;                      // 1024 padded keys / 32

  // Q fragments for this wave's 16 rows: 4 chunks of K=32 over E=128.
  v16bf qf[4];
  {
    const int qrow = row0 + wave * 16 + lm;
    #pragma unroll
    for (int c = 0; c < 4; c++) {
      const bf16* p = Q + (size_t)qrow * 1024 + ho + 32 * c;
      qf[c] = load_frag(p + 16 * hf, p + 16 * hf + 8);
    }
  }

  float mrow = NEG;
  float lrow = 0.f;
  v8f ofr[8];
  #pragma unroll
  for (int j = 0; j < 8; j++) ofr[j] = (v8f){};

#if HAS_TDM
  // Prologue: TDM-load chunk 0 into buffer 0 (wave 0 issues; EXEC ignored).
  if (wave == 0) {
    // K tile: 32 keys x 128 e; rows 256B + 16B pad -> LDS stride 272B.
    tdm_load_2d(lds_off(&Ks[0][0]), Kb + (size_t)0 * 1024 + ho,
                1024, 1024, 128, 32, 1024, /*pad_int 64DW*/5, /*pad 4DW*/3);
    // V tile: 128 e x 32 keys; rows 64B + 16B pad -> LDS stride 80B.
    tdm_load_2d(lds_off(&Vs[0][0]), Vt + (size_t)ho * 1024 + 0,
                1024, 1024, 32, 128, 1024, /*pad_int 16DW*/3, /*pad 4DW*/3);
  }
#endif

  for (int chunk = 0; chunk < NCHUNK; ++chunk) {
    const int buf = chunk & 1;
    const int s0  = chunk * 32;

#if HAS_TDM
    if (wave == 0) {
      if (chunk + 1 < NCHUNK) {
        const int sn = (chunk + 1) * 32;
        tdm_load_2d(lds_off(&Ks[buf ^ 1][0]), Kb + (size_t)sn * 1024 + ho,
                    1024, 1024, 128, 32, 1024, 5, 3);
        tdm_load_2d(lds_off(&Vs[buf ^ 1][0]), Vt + (size_t)ho * 1024 + sn,
                    1024, 1024, 32, 128, 1024, 3, 3);
        __builtin_amdgcn_s_wait_tensorcnt(2);   // current chunk complete
      } else {
        __builtin_amdgcn_s_wait_tensorcnt(0);
      }
    }
#else
    // Fallback staging (padded buffers: no guards).
    #pragma unroll
    for (int t = 0; t < 4; t++) {
      const int idx = (tid + 128 * t) * 8;
      const int key = idx >> 7;
      const int e   = idx & 127;
      u32x4 w = *(const u32x4*)(Kb + (size_t)(s0 + key) * 1024 + ho + e);
      *(u32x4*)(&Ks[buf][key * 136 + e]) = w;
    }
    #pragma unroll
    for (int t = 0; t < 4; t++) {
      const int idx = (tid + 128 * t) * 8;
      const int e   = idx >> 5;
      const int key = idx & 31;
      u32x4 w = *(const u32x4*)(Vt + (size_t)(ho + e) * 1024 + s0 + key);
      *(u32x4*)(&Vs[buf][e * 40 + key]) = w;
    }
#endif
    __syncthreads();

    // ---- scores: two 16x16 tiles (keys 0-15, 16-31), K=128 in 4 chunks ----
    const bf16* ks = Ks[buf];
    const bf16* vs = Vs[buf];
    v8f st0 = (v8f){}, st1 = (v8f){};
    #pragma unroll
    for (int c = 0; c < 4; c++) {
      const bf16* p0 = &ks[lm * 136 + 32 * c + 16 * hf];          // key tile 0
      const bf16* p1 = &ks[(16 + lm) * 136 + 32 * c + 16 * hf];   // key tile 1
      v16bf b0 = load_frag(p0, p0 + 8);
      v16bf b1 = load_frag(p1, p1 + 8);
      st0 = wmma_bf16(qf[c], b0, st0);
      st1 = wmma_bf16(qf[c], b1, st1);
    }

    // spill scores to wave-private LDS [16 rows][32 keys]
    float* pw = Pw[wave];
    #pragma unroll
    for (int r = 0; r < 8; r++) {
      pw[(r + 8 * hf) * 32 + lm]      = st0[r];
      pw[(r + 8 * hf) * 32 + 16 + lm] = st1[r];
    }
    lds_fence();

    // ---- online softmax: each lane owns row lm (dup across halves) ----
    float pv[32];
    float cmax = NEG;
    #pragma unroll
    for (int jj = 0; jj < 8; jj++) {
      v4f f = *(const v4f*)&pw[lm * 32 + 4 * jj];
      #pragma unroll
      for (int q2 = 0; q2 < 4; q2++) {
        const int j = 4 * jj + q2;
        const float v = (s0 + j < S) ? f[q2] * scale : NEG;
        pv[j] = v;
        cmax = fmaxf(cmax, v);
      }
    }
    const float mn    = fmaxf(mrow, cmax);
    const float alpha = __expf(mrow - mn);
    float sum = 0.f;
    #pragma unroll
    for (int j = 0; j < 32; j++) {
      float pe = __expf(pv[j] - mn);
      pe = (pv[j] == NEG) ? 0.f : pe;
      pv[j] = pe;
      sum += pe;
    }
    lrow = lrow * alpha + sum;
    mrow = mn;

    // write exp(P) as bf16: half 0 writes keys 0..15, half 1 keys 16..31
    {
      float sv[16];
      #pragma unroll
      for (int j = 0; j < 16; j++) sv[j] = hf ? pv[16 + j] : pv[j];
      u32x4 w0 = (u32x4){pack2(sv[0], sv[1]),  pack2(sv[2], sv[3]),
                         pack2(sv[4], sv[5]),  pack2(sv[6], sv[7])};
      u32x4 w1 = (u32x4){pack2(sv[8], sv[9]),  pack2(sv[10], sv[11]),
                         pack2(sv[12], sv[13]), pack2(sv[14], sv[15])};
      bf16* pb = Pb[wave];
      *(u32x4*)(&pb[lm * 32 + 16 * hf])     = w0;
      *(u32x4*)(&pb[lm * 32 + 16 * hf + 8]) = w1;
    }
    red[wave][lm] = alpha;       // both halves write identical value
    lds_fence();

    // rescale O accumulators by alpha of their C-layout rows
    #pragma unroll
    for (int r = 0; r < 8; r++) {
      const float a = red[wave][r + 8 * hf];
      #pragma unroll
      for (int j = 0; j < 8; j++) ofr[j][r] *= a;
    }

    // ---- P(16x32) x V(32x128): 8 WMMAs over the e-dimension ----
    const bf16* pp = &Pb[wave][lm * 32];
    v16bf ap = load_frag(pp + 16 * hf, pp + 16 * hf + 8);
    #pragma unroll
    for (int j = 0; j < 8; j++) {
      const bf16* p = &vs[(j * 16 + lm) * 40 + 16 * hf];
      v16bf bv2 = load_frag(p, p + 8);
      ofr[j] = wmma_bf16(ap, bv2, ofr[j]);
    }
    __syncthreads();
  }

  // ---- finalize: divide by l, store f32 [row][1024] ----
  red[wave][lm] = lrow;
  lds_fence();
  #pragma unroll
  for (int r = 0; r < 8; r++) {
    const float inv = 1.f / red[wave][r + 8 * hf];
    const int row = row0 + wave * 16 + r + 8 * hf;
    #pragma unroll
    for (int j = 0; j < 8; j++)
      Out[(size_t)row * 1024 + ho + j * 16 + lm] = ofr[j][r] * inv;
  }
}

// ---------------------------------------------------------------------------
// Launch: 5 kernels.  Workspace layout (bytes):
//   [0, 16777216)            q    bf16 [8192][1024]
//   [16777216, +2097152)     k    bf16 [1024][1024]   (rows 1000.. padded)
//   [18874368, +2097152)     vT   bf16 [1024][1024]   (cols 1000.. padded)
//   [20971520, +33554432)    attn f32  [8192][1024]   total ~54.5 MB
// ---------------------------------------------------------------------------
extern "C" void kernel_launch(void* const* d_in, const int* in_sizes, int n_in,
                              void* d_out, int out_size, void* d_ws, size_t ws_size,
                              hipStream_t stream)
{
  (void)in_sizes; (void)n_in; (void)out_size; (void)ws_size;
  const float* target = (const float*)d_in[0];
  const float* source = (const float*)d_in[1];
  const float* value  = (const float*)d_in[2];
  const float* Wq = (const float*)d_in[3];
  const float* bq = (const float*)d_in[4];
  const float* Wk = (const float*)d_in[5];
  const float* bk = (const float*)d_in[6];
  const float* Wv = (const float*)d_in[7];
  const float* bv = (const float*)d_in[8];
  const float* Wo = (const float*)d_in[9];
  const float* bo = (const float*)d_in[10];
  float* out = (float*)d_out;

  char* ws = (char*)d_ws;
  bf16*  qbf  = (bf16*)(ws);
  bf16*  kbf  = (bf16*)(ws + (size_t)16777216);
  bf16*  vtb  = (bf16*)(ws + (size_t)18874368);
  float* attn = (float*)(ws + (size_t)20971520);

  const dim3 blk(256);
  // Q = target @ Wq + bq               -> bf16 [8192,1024]
  gemm_bf16<1><<<dim3(8, 64), blk, 0, stream>>>(target, Wq, bq, qbf,
                                                8192, 8192, 1024, 1024, 0);
  // K = source @ Wk + bk, padded to 1024 rows -> bf16 [1024,1024]
  gemm_bf16<1><<<dim3(8, 8),  blk, 0, stream>>>(source, Wk, bk, kbf,
                                                1000, 1024, 1024, 4096, 0);
  // V = value @ Wv + bv, transposed, padded   -> bf16 [1024,1024]
  gemm_bf16<2><<<dim3(8, 8),  blk, 0, stream>>>(value,  Wv, bv, vtb,
                                                1000, 1024, 1024, 4096, 1024);
  // attention -> f32 [8192,1024]
  attn_fused<<<dim3(128, 8), dim3(128), 0, stream>>>(qbf, kbf, vtb, attn, 1000);
  // out = attn @ Wo + bo               -> f32 [8192,4096]
  gemm_bf16<0><<<dim3(32, 64), blk, 0, stream>>>(attn, Wo, bo, out,
                                                 8192, 8192, 4096, 1024, 0);
}
